// FourierCrossAttention_30562987278823
// MI455X (gfx1250) — matrix operand order repaired and checked by reading, not compiled, not run on hardware
//
#include <hip/hip_runtime.h>
#include <math.h>

// FourierCrossAttention for gfx1250 (MI455X): truncated-DFT-as-GEMM pipeline.
// All matmuls via v_wmma_f32_16x16x32_f16 (wave32 WMMA). All fragment data is
// stored "fragment-ready" ([tile][lane][16 halves]) so every fragment load is
// 2x b128; B-matrices are pre-swizzled in global memory by their producers.
// k_dft double-buffers its A tile (1 barrier/chunk); k_attn stages its four
// f16 planes with GLOBAL_LOAD_ASYNC_TO_LDS_B128 (ASYNCcnt).

typedef __attribute__((ext_vector_type(16))) _Float16 v16h;
typedef __attribute__((ext_vector_type(8)))  _Float16 v8h;
typedef __attribute__((ext_vector_type(8)))  float    v8f;

#define LQN   1024
#define NE    64
#define NO    64
#define NH    8
#define NB    32
#define NBH   (NB * NH)

// ---- fragment-position helpers (CDNA5 16x16x32 f16, wave32) ----------------
// B element (K5 in 0..31, N in tile): lane = (N&15) + 16*((K5>>3)&1),
// slot = (K5&7) + 8*(K5>>4)
__device__ __forceinline__ int bswz(int K5, int N) {
  return (((N & 15) + (((K5 >> 3) & 1) << 4)) << 4) + ((K5 & 7) + ((K5 >> 4) << 3));
}
// A element (M in tile, K5 in 0..31): lane = (M&15) + 16*((K5>>3)&1),
// slot = (K5&7) + 8*(K5>>4)
__device__ __forceinline__ int aswz(int M, int K5) {
  return (((M & 15) + (((K5 >> 3) & 1) << 4)) << 4) + ((K5 & 7) + ((K5 >> 4) << 3));
}

// Load a fragment from fragment-ready storage (global or LDS): 2x b128.
__device__ __forceinline__ v16h frag_ready(const _Float16* base, int lane) {
  const v8h* p = (const v8h*)(base + lane * 16);
  v8h lo = p[0], hi = p[1];
  return __builtin_shufflevector(lo, hi, 0, 1, 2, 3, 4, 5, 6, 7,
                                 8, 9, 10, 11, 12, 13, 14, 15);
}

// A fragment from row-major LDS tile (stride in halves, multiple of 8): 2x b128.
__device__ __forceinline__ v16h frag_a(const _Float16* p, int stride,
                                       int m0, int k0, int lane) {
  const _Float16* row = p + (m0 + (lane & 15)) * stride + k0 + ((lane >> 4) << 3);
  v8h lo = *(const v8h*)(row);
  v8h hi = *(const v8h*)(row + 16);
  return __builtin_shufflevector(lo, hi, 0, 1, 2, 3, 4, 5, 6, 7,
                                 8, 9, 10, 11, 12, 13, 14, 15);
}

__device__ __forceinline__ v8f wmma_f16(v16h a, v16h b, v8f c) {
  return __builtin_amdgcn_wmma_f32_16x16x32_f16(false, a, false, b,
                                                (short)0, c, false, false);
}

// D element r (0..7): M = r + 8*(lane>>4), N = lane%16
__device__ __forceinline__ int d_row(int r, int lane) { return r + ((lane >> 4) << 3); }
__device__ __forceinline__ int d_col(int lane)        { return lane & 15; }

// ---- CDNA5 async global->LDS copy (16 B per lane, ASYNCcnt-tracked) --------
__device__ __forceinline__ void async_copy16(void* lds_dst, const void* gsrc) {
  unsigned l = (unsigned)(uintptr_t)lds_dst;   // LDS aperture: offset in [31:0]
  asm volatile("global_load_async_to_lds_b128 %0, %1, off"
               :: "v"(l), "v"(gsrc) : "memory");
}
__device__ __forceinline__ void async_wait0() {
  asm volatile("s_wait_asynccnt 0x0" ::: "memory");
}

// ---- K1a: twiddle tables, written fragment-ready ---------------------------
// WtqF/WtkF (forward, B of k_dft): K = l (32 chunks), N = 2m+c (8 n-tiles):
//   [kc 32][nt 8][512].  WiF (inverse, B of k_idft): K = 2m+c (4 chunks),
//   N = l (64 n-tiles): [kc 4][nt 64][512].
__global__ __launch_bounds__(256) void k_tables(const int* __restrict__ idxq,
                                                const int* __restrict__ idxkv,
                                                _Float16* __restrict__ WtqF,
                                                _Float16* __restrict__ WtkF,
                                                _Float16* __restrict__ WiF) {
  int tid = blockIdx.x * 256 + threadIdx.x;   // 0 .. 1024*64-1
  int l = tid >> 6, m = tid & 63;
  const float w0 = 6.283185307179586f / (float)LQN;
  int fq = idxq[m], fk = idxkv[m];
  float sq, cq, sk, ck;
  sincosf((float)((l * fq) & (LQN - 1)) * w0, &sq, &cq);
  sincosf((float)((l * fk) & (LQN - 1)) * w0, &sk, &ck);
  // forward: element (K=l, N=2m+c)
  int kc = l >> 5, K5 = l & 31, nt = (2 * m) >> 4;
  int base = (kc * 8 + nt) * 512;
  WtqF[base + bswz(K5, 2 * m)]     = (_Float16)cq;
  WtqF[base + bswz(K5, 2 * m + 1)] = (_Float16)(-sq);
  WtkF[base + bswz(K5, 2 * m)]     = (_Float16)ck;
  WtkF[base + bswz(K5, 2 * m + 1)] = (_Float16)(-sk);
  // inverse: element (K=2m+c, N=l), with irfft mode coefficient folded in
  float fm = (fq == 0 || fq == LQN / 2) ? 1.0f : 2.0f;
  int ntl = l >> 4;
  int K0 = 2 * m, kci = K0 >> 5;
  WiF[(kci * 64 + ntl) * 512 + bswz(K0 & 31, l)]       = (_Float16)(fm * cq);
  WiF[(kci * 64 + ntl) * 512 + bswz((K0 + 1) & 31, l)] = (_Float16)(-fm * sq);
}

// ---- K1b: weight transpose (H,E,O,X) f32 -> row-major [h][x][e][o] f16 -----
__global__ __launch_bounds__(256) void k_wprep(const float* __restrict__ wr,
                                               const float* __restrict__ wi,
                                               _Float16* __restrict__ WwTR,
                                               _Float16* __restrict__ WwTI) {
  int he = blockIdx.x;                       // h*64 + e
  __shared__ __align__(16) _Float16 tR[64 * 72], tI[64 * 72];
  const float* pr = wr + (size_t)he * 4096;  // [o][x], x fastest
  const float* pi = wi + (size_t)he * 4096;
#pragma unroll
  for (int i = 0; i < 16; ++i) {
    int idx = threadIdx.x + i * 256;
    int o = idx >> 6, x = idx & 63;
    tR[x * 72 + o] = (_Float16)pr[idx];
    tI[x * 72 + o] = (_Float16)pi[idx];
  }
  __syncthreads();
  int h = he >> 6, e = he & 63;
  size_t obase = (size_t)h * 262144 + (size_t)e * 64;  // [h][x][e][o]
#pragma unroll
  for (int i = 0; i < 2; ++i) {
    int v = threadIdx.x + i * 256;           // 0..511
    int x = v >> 3, og = (v & 7) << 3;
    *(v8h*)&WwTR[obase + (size_t)x * 4096 + og] = *(const v8h*)&tR[x * 72 + og];
    *(v8h*)&WwTI[obase + (size_t)x * 4096 + og] = *(const v8h*)&tI[x * 72 + og];
  }
}

// ---- K2: forward truncated DFT as GEMM ------------------------------------
// Per (which,b,h): C[64e x 128(2m+c)] = A[64e x 1024l] * Wt[1024l x 128].
// Double-buffered A tile: one barrier per K-chunk, loads overlap WMMA.
// q -> Xq row-major [bh][m][e]; k -> XkA row-major [bh][e][m] (A of phase 2)
// plus XkB fragment-ready [bh][kc 2][nt 4][512] (B of phase 1).
__global__ __launch_bounds__(256) void k_dft(const float* __restrict__ q,
                                             const float* __restrict__ k,
                                             const _Float16* __restrict__ WtqF,
                                             const _Float16* __restrict__ WtkF,
                                             _Float16* __restrict__ XqR,
                                             _Float16* __restrict__ XqI,
                                             _Float16* __restrict__ XkR,
                                             _Float16* __restrict__ XkI,
                                             _Float16* __restrict__ XkBR,
                                             _Float16* __restrict__ XkBI) {
  const int which = blockIdx.x >> 8;   // 0 = q, 1 = k
  const int bh = blockIdx.x & 255;
  const int bi = bh >> 3, h = bh & 7;
  const int t = threadIdx.x, lane = t & 31, wv = t >> 5;
  const float* src = (which == 0) ? q : k;
  const _Float16* WtF = (which == 0) ? WtqF : WtkF;

  __shared__ __align__(16) _Float16 ldsA[2][64 * 40];   // [e][l-chunk 32] x2

  const int mt = wv >> 1, nt0 = (wv & 1) * 4;
  v8f acc[4] = {};

  const int v0 = t, v1 = t + 256;   // two float4 slots per thread per chunk
  const int l_0 = v0 >> 4, e4_0 = (v0 & 15) << 2;
  const int l_1 = v1 >> 4, e4_1 = (v1 & 15) << 2;

  auto gsrc = [&](int l0, int l, int e4) {
    return (const float4*)&src[((size_t)(bi * LQN + l0 + l) * NH + h) * NE + e4];
  };
  auto stash = [&](int buf, int l, int e4, float4 f) {
    _Float16* A = ldsA[buf];
    A[(e4 + 0) * 40 + l] = (_Float16)f.x;
    A[(e4 + 1) * 40 + l] = (_Float16)f.y;
    A[(e4 + 2) * 40 + l] = (_Float16)f.z;
    A[(e4 + 3) * 40 + l] = (_Float16)f.w;
  };

  {
    float4 f0 = *gsrc(0, l_0, e4_0);
    float4 f1 = *gsrc(0, l_1, e4_1);
    stash(0, l_0, e4_0, f0);
    stash(0, l_1, e4_1, f1);
  }
  __syncthreads();

  for (int it = 0; it < 32; ++it) {
    const int buf = it & 1;
    float4 f0, f1;
    if (it < 31) {                    // issue next-chunk loads before math
      f0 = *gsrc((it + 1) * 32, l_0, e4_0);
      f1 = *gsrc((it + 1) * 32, l_1, e4_1);
    }
    v16h a = frag_a(ldsA[buf], 40, mt * 16, 0, lane);
#pragma unroll
    for (int j = 0; j < 4; ++j) {
      v16h b = frag_ready(WtF + (size_t)(it * 8 + nt0 + j) * 512, lane);
      acc[j] = wmma_f16(a, b, acc[j]);
    }
    if (it < 31) {
      stash(buf ^ 1, l_0, e4_0, f0);
      stash(buf ^ 1, l_1, e4_1, f1);
    }
    __syncthreads();
  }
#pragma unroll
  for (int j = 0; j < 4; ++j)
#pragma unroll
    for (int r = 0; r < 8; ++r) {
      int e = mt * 16 + d_row(r, lane);
      int col = (nt0 + j) * 16 + d_col(lane);
      int m = col >> 1, cpart = col & 1;
      _Float16 v = (_Float16)acc[j][r];
      if (which == 0) {
        _Float16* dst = cpart ? XqI : XqR;
        dst[(size_t)(bh * 64 + m) * 64 + e] = v;            // [bh][m][e]
      } else {
        _Float16* dst = cpart ? XkI : XkR;
        dst[(size_t)(bh * 64 + e) * 64 + m] = v;            // [bh][e][m]
        _Float16* dstB = cpart ? XkBI : XkBR;               // B-frag layout
        dstB[(size_t)bh * 4096 + ((e >> 5) * 4 + (m >> 4)) * 512 +
             bswz(e & 31, m)] = v;
      }
    }
}

// ---- K3: scores + tanh + t, fused per (b,h) -------------------------------
// S[x,y] = sum_e Xq[e,x]*Xk[e,y] (complex), tanh per component,
// t[e,x] = sum_y Xk[e,y]*S[x,y].  t written A-fragment-ready for k_proj:
// [h][x][kc 2][bt 2][512].
__global__ __launch_bounds__(256) void k_attn(const _Float16* __restrict__ XqR,
                                              const _Float16* __restrict__ XqI,
                                              const _Float16* __restrict__ XkR,
                                              const _Float16* __restrict__ XkI,
                                              const _Float16* __restrict__ XkBR,
                                              const _Float16* __restrict__ XkBI,
                                              _Float16* __restrict__ tFR,
                                              _Float16* __restrict__ tFI) {
  const int bh = blockIdx.x;
  const int bi = bh >> 3, h = bh & 7;
  const int t = threadIdx.x, lane = t & 31, wv = t >> 5;
  __shared__ __align__(16) _Float16 qR[64 * 40], qI[64 * 40];  // [x][e]
  __shared__ __align__(16) _Float16 kR[64 * 40], kI[64 * 40];  // [e][y]
  __shared__ __align__(16) _Float16 sFR[4096], sFI[4096];      // tanh(S)^T, B-frag
  const size_t base = (size_t)bh * 4096;
  // async global->LDS staging of the four planes (GLOBAL_LOAD_ASYNC_TO_LDS_B128)
#pragma unroll
  for (int i = 0; i < 2; ++i) {
    int v = t + i * 256;                    // 0..511
    int r = v >> 3, g = (v & 7) << 3;
    async_copy16(&qR[r * 40 + g], &XqR[base + r * 64 + g]);
    async_copy16(&qI[r * 40 + g], &XqI[base + r * 64 + g]);
    async_copy16(&kR[r * 40 + g], &XkR[base + r * 64 + g]);
    async_copy16(&kI[r * 40 + g], &XkI[base + r * 64 + g]);
  }
  async_wait0();
  __syncthreads();
  // phase 1: S (16 complex tiles, 2 per wave); write tanh(S)^T B-frag to LDS
#pragma unroll
  for (int p = 0; p < 2; ++p) {
    int pos = wv * 2 + p;
    int xt = pos >> 2, yt = pos & 3;
    v8f rr = {}, ii = {}, ri = {}, ir = {};
#pragma unroll
    for (int e0 = 0; e0 < 64; e0 += 32) {
      int kc = e0 >> 5;
      v16h aR = frag_a(qR, 40, xt * 16, e0, lane);
      v16h aI = frag_a(qI, 40, xt * 16, e0, lane);
      v16h bR = frag_ready(XkBR + base + (kc * 4 + yt) * 512, lane);
      v16h bI = frag_ready(XkBI + base + (kc * 4 + yt) * 512, lane);
      rr = wmma_f16(aR, bR, rr); ii = wmma_f16(aI, bI, ii);
      ri = wmma_f16(aR, bI, ri); ir = wmma_f16(aI, bR, ir);
    }
#pragma unroll
    for (int r = 0; r < 8; ++r) {
      int x = xt * 16 + d_row(r, lane);
      int y = yt * 16 + d_col(lane);
      int off = ((y >> 5) * 4 + xt) * 512 + bswz(y & 31, x);  // (K=y, N=x)
      sFR[off] = (_Float16)tanhf(rr[r] - ii[r]);
      sFI[off] = (_Float16)tanhf(ri[r] + ir[r]);
    }
  }
  __syncthreads();
  // phase 2: t = Xk (A = [e][y]) x tanh(S)^T (B = [y][x]), complex
#pragma unroll
  for (int p = 0; p < 2; ++p) {
    int pos = wv * 2 + p;
    int et = pos >> 2, xt = pos & 3;
    v8f rr = {}, ii = {}, ri = {}, ir = {};
#pragma unroll
    for (int y0 = 0; y0 < 64; y0 += 32) {
      int kc = y0 >> 5;
      v16h aR = frag_a(kR, 40, et * 16, y0, lane);
      v16h aI = frag_a(kI, 40, et * 16, y0, lane);
      v16h bR = frag_ready(&sFR[(kc * 4 + xt) * 512], lane);
      v16h bI = frag_ready(&sFI[(kc * 4 + xt) * 512], lane);
      rr = wmma_f16(aR, bR, rr); ii = wmma_f16(aI, bI, ii);
      ri = wmma_f16(aR, bI, ri); ir = wmma_f16(aI, bR, ir);
    }
#pragma unroll
    for (int r = 0; r < 8; ++r) {
      int e = et * 16 + d_row(r, lane);
      int x = xt * 16 + d_col(lane);
      size_t off = ((size_t)(h * 64 + x) * 4 + (e >> 5) * 2 + (bi >> 4)) * 512 +
                   aswz(bi, e & 31);           // A element (M=b, K=e)
      tFR[off] = (_Float16)(rr[r] - ii[r]);
      tFI[off] = (_Float16)(ri[r] + ir[r]);
    }
  }
}

// ---- K4: per-(h,x) projection: XW[b,o] = sum_e t[b,e]*w[e,o] (complex) ----
// A from tF (fragment-ready global); B staged swizzled into LDS from WwT.
// Output written A-fragment-ready for k_idft: [rt 1024][kc 4][512].
__global__ __launch_bounds__(256) void k_proj(const _Float16* __restrict__ tFR,
                                              const _Float16* __restrict__ tFI,
                                              const _Float16* __restrict__ WwTR,
                                              const _Float16* __restrict__ WwTI,
                                              _Float16* __restrict__ XWf) {
  const int h = blockIdx.x >> 6, x = blockIdx.x & 63;
  const int t = threadIdx.x, lane = t & 31, wv = t >> 5;
  __shared__ __align__(16) _Float16 lwR[4096], lwI[4096];  // B-frag [kc 2][nt 4][512]
  const size_t wbase = (size_t)(h * 64 + x) * 4096;        // WwT slab [e][o]
#pragma unroll
  for (int i = 0; i < 2; ++i) {
    int v = t + i * 256;                    // 0..511
    int e = v >> 3, og = (v & 7) << 3;
    v8h r8 = *(const v8h*)&WwTR[wbase + e * 64 + og];
    v8h i8 = *(const v8h*)&WwTI[wbase + e * 64 + og];
    int kc = e >> 5, K5 = e & 31;
    int bb = (kc * 4 + (og >> 4)) * 512;
#pragma unroll
    for (int j = 0; j < 8; ++j) {
      int off = bb + bswz(K5, og + j);
      lwR[off] = r8[j];
      lwI[off] = i8[j];
    }
  }
  __syncthreads();
  const int bt = wv >> 2, ot = wv & 3;      // 8 waves = 2 b-tiles x 4 o-tiles
  const size_t tb = (size_t)(h * 64 + x) * 4;
  v8f rr = {}, ii = {}, ri = {}, ir = {};
#pragma unroll
  for (int e0 = 0; e0 < 64; e0 += 32) {
    int kc = e0 >> 5;
    v16h aR = frag_ready(tFR + (tb + kc * 2 + bt) * 512, lane);
    v16h aI = frag_ready(tFI + (tb + kc * 2 + bt) * 512, lane);
    v16h bR = frag_ready(&lwR[(kc * 4 + ot) * 512], lane);
    v16h bI = frag_ready(&lwI[(kc * 4 + ot) * 512], lane);
    rr = wmma_f16(aR, bR, rr); ii = wmma_f16(aI, bI, ii);
    ri = wmma_f16(aR, bI, ri); ir = wmma_f16(aI, bR, ir);
  }
#pragma unroll
  for (int r = 0; r < 8; ++r) {
    int b = bt * 16 + d_row(r, lane);
    int o = ot * 16 + d_col(lane);
    int row = (b * NH + h) * NO + o;
    int K0 = 2 * x;                          // K = 2x+c, kc = x>>4
    size_t ob = ((size_t)(row >> 4) * 4 + (x >> 4)) * 512;
    XWf[ob + aswz(row, K0 & 31)]       = (_Float16)(rr[r] - ii[r]);
    XWf[ob + aswz(row, (K0 + 1) & 31)] = (_Float16)(ri[r] + ir[r]);
  }
}

// ---- K5: inverse DFT GEMM (no LDS): out[row,l] = sc * sum_K XW*Wi ---------
__global__ __launch_bounds__(256) void k_idft(const _Float16* __restrict__ XWf,
                                              const _Float16* __restrict__ WiF,
                                              float* __restrict__ out) {
  const int rb = blockIdx.x >> 3;     // 256 row blocks of 64
  const int cb = blockIdx.x & 7;      // 8 col blocks of 128
  const int t = threadIdx.x, lane = t & 31, wv = t >> 5;
  const int mt = wv >> 1, nt0 = (wv & 1) * 4;
  const int rt = rb * 4 + mt;         // 16-row tile index
  v8f acc[4] = {};
#pragma unroll
  for (int kc = 0; kc < 4; ++kc) {
    v16h a = frag_ready(XWf + ((size_t)rt * 4 + kc) * 512, lane);
#pragma unroll
    for (int j = 0; j < 4; ++j) {
      v16h b = frag_ready(WiF + ((size_t)kc * 64 + cb * 8 + nt0 + j) * 512, lane);
      acc[j] = wmma_f16(a, b, acc[j]);
    }
  }
  const float sc = 1.0f / (1024.0f * 262144.0f);  // 1/L * 1/(IN_CH*OUT_CH)
#pragma unroll
  for (int j = 0; j < 4; ++j)
#pragma unroll
    for (int r = 0; r < 8; ++r) {
      int row = rt * 16 + d_row(r, lane);
      int col = cb * 128 + (nt0 + j) * 16 + d_col(lane);
      out[(size_t)row * LQN + col] = acc[j][r] * sc;
    }
}

// ---- host-side launcher ----------------------------------------------------
extern "C" void kernel_launch(void* const* d_in, const int* in_sizes, int n_in,
                              void* d_out, int out_size, void* d_ws, size_t ws_size,
                              hipStream_t stream) {
  const float* q  = (const float*)d_in[0];
  const float* k  = (const float*)d_in[1];
  // d_in[2] = v, d_in[3] = mask: unused by the reference computation
  const float* wr = (const float*)d_in[4];
  const float* wi = (const float*)d_in[5];
  const int* idxq  = (const int*)d_in[6];
  const int* idxkv = (const int*)d_in[7];
  float* out = (float*)d_out;

  size_t off = 0;
  auto carve = [&](size_t bytes) {
    void* p = (char*)d_ws + off;
    off += (bytes + 255) & ~(size_t)255;
    return p;
  };
  _Float16* WtqF = (_Float16*)carve((size_t)131072 * 2);   // 256 KB each
  _Float16* WtkF = (_Float16*)carve((size_t)131072 * 2);
  _Float16* WiF  = (_Float16*)carve((size_t)131072 * 2);
  _Float16* WwTR = (_Float16*)carve((size_t)2097152 * 2);  // 4 MB each
  _Float16* WwTI = (_Float16*)carve((size_t)2097152 * 2);
  _Float16* XqR  = (_Float16*)carve((size_t)1048576 * 2);  // 2 MB each
  _Float16* XqI  = (_Float16*)carve((size_t)1048576 * 2);
  _Float16* XkR  = (_Float16*)carve((size_t)1048576 * 2);
  _Float16* XkI  = (_Float16*)carve((size_t)1048576 * 2);
  _Float16* XkBR = (_Float16*)carve((size_t)1048576 * 2);
  _Float16* XkBI = (_Float16*)carve((size_t)1048576 * 2);
  _Float16* tFR  = (_Float16*)carve((size_t)1048576 * 2);
  _Float16* tFI  = (_Float16*)carve((size_t)1048576 * 2);
  _Float16* XWf  = (_Float16*)carve((size_t)2097152 * 2);  // 4 MB

  k_tables<<<256, 256, 0, stream>>>(idxq, idxkv, WtqF, WtkF, WiF);
  k_wprep <<<NH * NE, 256, 0, stream>>>(wr, wi, WwTR, WwTI);
  k_dft   <<<2 * NBH, 256, 0, stream>>>(q, k, WtqF, WtkF,
                                        XqR, XqI, XkR, XkI, XkBR, XkBI);
  k_attn  <<<NBH, 256, 0, stream>>>(XqR, XqI, XkR, XkI, XkBR, XkBI, tFR, tFI);
  k_proj  <<<NH * 64, 256, 0, stream>>>(tFR, tFI, WwTR, WwTI, XWf);
  k_idft  <<<256 * 8, 256, 0, stream>>>(XWf, WiF, out);
}
